// DilateAttention_41661182771556
// MI455X (gfx1250) — matrix operand order repaired and checked by reading, not compiled, not run on hardware
//
#include <hip/hip_runtime.h>

// CDNA5 / gfx1250 dilated (3x3) neighborhood attention, bf16 WMMA formulation.
// q,k,v: [B=32, d=64, H=64, W=64] f32 -> out same shape f32.

typedef __attribute__((ext_vector_type(16))) __bf16 v16bf;
typedef __attribute__((ext_vector_type(8)))  float  v8f;

#define WMMA_BF16(a, b, c) \
  __builtin_amdgcn_wmma_f32_16x16x32_bf16(false, (a), false, (b), (short)0, (c), false, false)

// Same-wave LDS staging: DS ops are in-order per wave in HW; we only need a
// compiler-level memory fence so stores/loads are not reordered.
#define FENCE() asm volatile("" ::: "memory")

constexpr int Bb = 32, Dd = 64, Hh = 64, Ww = 64;
constexpr int HT    = 4;        // image rows per workgroup
constexpr int KROWS = HT + 2;   // +halo rows
constexpr int CCOLS = Ww + 2;   // +halo cols (c = w+1, c in [0,65])
constexpr int KSTR  = Dd + 2;   // d-stride pad for Klds (bank-conflict-free stores)
constexpr int SSTR  = 17;       // stage row stride (conflict-free diagonal reads)
constexpr float SCALE_F = 0.125f;   // 64^-0.5

__global__ __launch_bounds__(256) void dilate_attn_wmma(
    const float* __restrict__ qg, const float* __restrict__ kg,
    const float* __restrict__ vg, float* __restrict__ og)
{
  __shared__ __bf16 Klds[KROWS][CCOLS][KSTR];   // [row][col][d] : contiguous d for QK B-op
  __shared__ __bf16 Vlds[KROWS][Dd][CCOLS];     // [row][d][col] : contiguous col for AV B-op
  __shared__ float  Stage[8][16 * SSTR];        // per-wave C staging
  __shared__ __bf16 Aband[8][3 * 512];          // per-wave banded attn A-matrices (3 of 16x32)

  const int tid  = threadIdx.x;
  const int wave = tid >> 5;
  const int lane = tid & 31;
  const int m    = lane & 15;   // pixel-in-tile / N-column index
  const int hi   = lane >> 4;   // half-wave id

  const int b    = blockIdx.x >> 4;            // 32 batches
  const int h0   = (blockIdx.x & 15) * HT;     // 16 row-tiles
  const int base = b * Dd * Hh * Ww;

  // ---------------- Phase 1: stage K,V (bf16, zero-padded halo) into LDS ----------------
  for (int idx = tid; idx < KROWS * CCOLS * Dd; idx += 256) {
    int c  = idx % CCOLS;
    int r2 = idx / CCOLS;
    int d  = r2 & 63;
    int rr = r2 >> 6;
    int row = h0 - 1 + rr;
    int w   = c - 1;
    bool in = (row >= 0) && (row < Hh) && (w >= 0) && (w < Ww);
    float kv = 0.f, vv = 0.f;
    if (in) {
      int gidx = base + (d * Hh + row) * Ww + w;
      kv = kg[gidx];
      vv = vg[gidx];
    }
    Klds[rr][c][d] = (__bf16)kv;
    Vlds[rr][d][c] = (__bf16)vv;
  }
  __syncthreads();

  // ---------------- Phase 2: per-wave 16-pixel tiles ----------------
  // 16 tiles per WG (4 rows x 4 col-tiles); 8 waves x 2 tiles. Uniform trip count
  // across waves, so no block barriers needed past this point.
  float* St = &Stage[wave][0];
  __bf16* Ab = &Aband[wave][0];

  for (int it = 0; it < 2; ++it) {
    const int t  = wave * 2 + it;
    const int tr = t >> 2;             // row within tile-block (0..3)
    const int w0 = (t & 3) << 4;       // tile base column
    const int h  = h0 + tr;

    // ---- A operand: Q[16px x 64d] as two bf16 16x32 chunks, read direct from HBM.
    // A layout: lane m holds row m; K element e -> hi*16+e (contiguous).
    v16bf aq0, aq1;
#pragma unroll
    for (int e = 0; e < 16; ++e) {
      int dk = hi * 16 + e;
      aq0[e] = (__bf16)qg[base + (dk * Hh + h) * Ww + w0 + m];
      aq1[e] = (__bf16)qg[base + ((dk + 32) * Hh + h) * Ww + w0 + m];
    }

    // ---- QK: per row-offset di, two shifted products give all 3 dj diagonals.
    float sc[9];
#pragma unroll
    for (int i = 0; i < 9; ++i) sc[i] = 0.f;

#pragma unroll
    for (int di = 0; di < 3; ++di) {
      const int rblk = tr + di;                       // LDS halo row
      // B layout: lane holds N-col j = m; K element e -> d = d0 + hi*16 + e (contiguous).
      const __bf16* kp0 = &Klds[rblk][w0 + m][hi * 16];       // shift 0: window col w0-1+j
      const __bf16* kp1 = &Klds[rblk][w0 + m + 2][hi * 16];   // shift 1: window col w0+1+j
      v16bf b00, b01, b10, b11;
#pragma unroll
      for (int e = 0; e < 16; ++e) {
        b00[e] = kp0[e];       b01[e] = kp0[32 + e];
        b10[e] = kp1[e];       b11[e] = kp1[32 + e];
      }
      v8f c0 = {0.f,0.f,0.f,0.f,0.f,0.f,0.f,0.f};
      v8f c1 = {0.f,0.f,0.f,0.f,0.f,0.f,0.f,0.f};
      c0 = WMMA_BF16(aq0, b00, c0);  c0 = WMMA_BF16(aq1, b01, c0);
      c1 = WMMA_BF16(aq0, b10, c1);  c1 = WMMA_BF16(aq1, b11, c1);

      // Stage C (lane: N=m, rows M=hi*8+e) then pull diagonals per pixel-lane.
#pragma unroll
      for (int e = 0; e < 8; ++e) St[(hi * 8 + e) * SSTR + m] = c0[e];
      FENCE();
      if (hi == 0) {
        sc[di * 3 + 0] = St[m * SSTR + m];                         // dj=-1 : C0[m][m]
        if (m < 15) sc[di * 3 + 1] = St[m * SSTR + m + 1];         // dj= 0 : C0[m][m+1]
      }
      FENCE();
#pragma unroll
      for (int e = 0; e < 8; ++e) St[(hi * 8 + e) * SSTR + m] = c1[e];
      FENCE();
      if (hi == 0) {
        sc[di * 3 + 2] = St[m * SSTR + m];                         // dj=+1 : C1[m][m]
        if (m == 15) sc[di * 3 + 1] = St[15 * SSTR + 14];          // spilled dj=0
      }
      FENCE();
    }

    // ---- softmax over the 9-window (lanes 0-15 own pixels), f32 VALU.
    float w9[9];
#pragma unroll
    for (int i = 0; i < 9; ++i) w9[i] = 0.f;
    if (hi == 0) {
      float mx = sc[0];
#pragma unroll
      for (int i = 1; i < 9; ++i) mx = fmaxf(mx, sc[i]);
      float sum = 0.f;
#pragma unroll
      for (int i = 0; i < 9; ++i) { float e_ = __expf((sc[i] - mx) * SCALE_F); w9[i] = e_; sum += e_; }
      float inv = 1.f / sum;
#pragma unroll
      for (int i = 0; i < 9; ++i) w9[i] *= inv;
    }

    // ---- Build banded A-matrices: 6 (shift,di) 16x16 bands packed pairwise into
    // three 16x32 bf16 A operands. blk = s*3+di; matrix = blk/2, K-half = blk&1.
    for (int i = lane; i < 3 * 512; i += 32) Ab[i] = (__bf16)0.f;
    FENCE();
    if (hi == 0) {
#pragma unroll
      for (int di = 0; di < 3; ++di)
#pragma unroll
        for (int dj = 0; dj < 3; ++dj) {
          float wv = w9[di * 3 + dj];
          int j = m + dj;                  // shift0 target col (dj_actual = dj-1 -> j = m+dj)
          int blk = di;                    // s = 0
          if (j > 15) { j -= 2; blk = 3 + di; }   // spill to shift1 block
          Ab[(blk >> 1) * 512 + m * 32 + (blk & 1) * 16 + j] = (__bf16)wv;
        }
    }
    FENCE();

    // ---- AV: out[16px x 64d] = sum of 3 banded WMMAs per 16-wide d-chunk.
#pragma unroll
    for (int dbase = 0; dbase < 64; dbase += 16) {
      v8f co = {0.f,0.f,0.f,0.f,0.f,0.f,0.f,0.f};
#pragma unroll
      for (int i = 0; i < 3; ++i) {
        // A operand (band matrix i), same A layout as Q.
        v16bf av;
#pragma unroll
        for (int e = 0; e < 16; ++e) {
          int kk = hi * 16 + e;
          av[e] = Ab[i * 512 + m * 32 + kk];
        }
        // B operand: V rows for this lane's K-half block. blk = 2i+hi -> (s, di).
        int blk = 2 * i + hi;
        int s  = blk / 3;
        int di = blk % 3;
        const __bf16* vp = &Vlds[tr + di][dbase + m][w0 + 2 * s];   // contiguous cols j=e
        v16bf bv;
#pragma unroll
        for (int e = 0; e < 16; ++e) bv[e] = vp[e];
        co = WMMA_BF16(av, bv, co);
      }
      // C layout: lane N = d-col = dbase+m, rows M = pixel = hi*8+e -> 8 contiguous w.
      float* op = og + (size_t)((b * Dd + dbase + m) * Hh + h) * Ww + w0 + hi * 8;
#pragma unroll
      for (int e = 0; e < 8; ++e) op[e] = co[e];
    }
  }
}

extern "C" void kernel_launch(void* const* d_in, const int* in_sizes, int n_in,
                              void* d_out, int out_size, void* d_ws, size_t ws_size,
                              hipStream_t stream) {
  const float* q = (const float*)d_in[0];
  const float* k = (const float*)d_in[1];
  const float* v = (const float*)d_in[2];
  float* out = (float*)d_out;
  dim3 grid(Bb * (Hh / HT));   // 32 batches x 16 row-tiles = 512 workgroups
  dim3 block(256);             // 8 wave32
  hipLaunchKernelGGL(dilate_attn_wmma, grid, block, 0, stream, q, k, v, out);
}
